// TriangleAttentionStartingNode_2199023255612
// MI455X (gfx1250) — matrix operand order repaired and checked by reading, not compiled, not run on hardware
//
#include <hip/hip_runtime.h>

// TriangleAttentionStartingNode for MI455X (gfx1250, wave32, WMMA bf16).
// B=1, N=256, C=128, HC=32, H=4. All GEMMs via v_wmma_f32_16x16x32_bf16.

#define NRES   256
#define CCH    128
#define HC_    32
#define NH_    4
#define DM_    128
#define EPSV   1e-5f

typedef __attribute__((ext_vector_type(16))) __bf16 v16bf;
typedef __attribute__((ext_vector_type(8)))  __bf16 v8bf;
typedef __attribute__((ext_vector_type(8)))  float  v8f;

static __device__ __forceinline__ unsigned short f2bf(float f) {
  unsigned int u = __float_as_uint(f);
  u += 0x7FFFu + ((u >> 16) & 1u);          // round-to-nearest-even
  return (unsigned short)(u >> 16);
}
static __device__ __forceinline__ float bf2f(unsigned short h) {
  return __uint_as_float(((unsigned int)h) << 16);
}

// A-matrix 16x32 bf16 fragment from row-major LDS [rows][ld].
// Lanes 0-15: M=lane, K = k0+0..7 (v0..3) and k0+16..23 (v4..7)
// Lanes 16-31: M=lane-16, K = k0+8..15 and k0+24..31
static __device__ __forceinline__ v16bf frag_a(const unsigned short* base,
                                               int row0, int k0, int ld, int lane) {
  const int m  = lane & 15;
  const int kb = (lane >> 4) << 3;          // 0 or 8
  const unsigned short* p = base + (row0 + m) * ld + k0 + kb;
  v16bf r;
  ((v8bf*)&r)[0] = *(const v8bf*)(p);       // ds_load_b128
  ((v8bf*)&r)[1] = *(const v8bf*)(p + 16);  // ds_load_b128
  return r;
}

// B-matrix 32x16 bf16 fragment from *transposed* LDS storage Bt[n][k] (ld = k-stride).
// Lanes 0-15: N=lane, K = k0+0..7 and k0+16..23 ; lanes 16-31: N=lane-16, K = k0+8..15 and k0+24..31
static __device__ __forceinline__ v16bf frag_bT(const unsigned short* base,
                                                int n0, int k0, int ld, int lane) {
  const int n  = lane & 15;
  const int ko = (lane >> 4) << 3;          // 0 or 8
  const unsigned short* p = base + (n0 + n) * ld + k0 + ko;
  v16bf r;
  ((v8bf*)&r)[0] = *(const v8bf*)(p);
  ((v8bf*)&r)[1] = *(const v8bf*)(p + 16);
  return r;
}

#define WMMA_BF16(a, b, c) \
  __builtin_amdgcn_wmma_f32_16x16x32_bf16(false, (a), false, (b), (short)0, (c), false, false)

static __device__ __forceinline__ float wave_max(float v) {
#pragma unroll
  for (int m = 16; m >= 1; m >>= 1) v = fmaxf(v, __shfl_xor(v, m, 32));
  return v;
}
static __device__ __forceinline__ float wave_sum(float v) {
#pragma unroll
  for (int m = 16; m >= 1; m >>= 1) v += __shfl_xor(v, m, 32);
  return v;
}

// ---------------------------------------------------------------------------
// Kernel 1: LayerNorm + fused projections q,k,v,g (+bg, sigmoid) and b.
//   A-tile: 64 rows x 128 ch, bf16 in LDS (ld=136).
//   Weights: combined 128 x 528 (q|k|v|g|b+pad), staged transposed [col][k].
//   q,k,v -> bf16 [i][h][j][c]; g -> bf16 [m][f] (post-sigmoid); b -> f32 [h][j][i].
// ---------------------------------------------------------------------------
#define K1_LD   136
#define K1_NCOL 528

__global__ void __launch_bounds__(256) k1_ln_proj(
    const float* __restrict__ x, const float* __restrict__ gamma, const float* __restrict__ beta,
    const float* __restrict__ Wq, const float* __restrict__ Wk, const float* __restrict__ Wv,
    const float* __restrict__ Wb, const float* __restrict__ Wg, const float* __restrict__ bg,
    unsigned short* __restrict__ qT, unsigned short* __restrict__ kT, unsigned short* __restrict__ vT,
    unsigned short* __restrict__ gT, float* __restrict__ bT)
{
  extern __shared__ unsigned char smem_raw[];
  unsigned short* At = (unsigned short*)smem_raw;     // 64 x 136 bf16
  unsigned short* Wt = At + 64 * K1_LD;               // 528 x 136 bf16, [col][k]

  const int tid = threadIdx.x, lane = tid & 31, wid = tid >> 5;
  const int m0 = blockIdx.x * 64;

  // Stage combined weights, transposed, converted to bf16 (coalesced global reads).
  for (int idx = tid; idx < CCH * K1_NCOL; idx += 256) {
    const int k = idx / K1_NCOL;
    const int f = idx - k * K1_NCOL;
    float w;
    if      (f < 128) w = Wq[k * 128 + f];
    else if (f < 256) w = Wk[k * 128 + (f - 128)];
    else if (f < 384) w = Wv[k * 128 + (f - 256)];
    else if (f < 512) w = Wg[k * 128 + (f - 384)];
    else if (f < 516) w = Wb[k * 4 + (f - 512)];
    else              w = 0.0f;
    Wt[f * K1_LD + k] = f2bf(w);
  }

  // LayerNorm: one wave per row (8 rows/wave), lane covers 4 channels.
  const float4 ga = *(const float4*)(gamma + lane * 4);
  const float4 be = *(const float4*)(beta  + lane * 4);
#pragma unroll
  for (int r8 = 0; r8 < 8; ++r8) {
    const int row = wid * 8 + r8;
    const float4 xv = *(const float4*)(x + (size_t)(m0 + row) * CCH + lane * 4);
    float s  = xv.x + xv.y + xv.z + xv.w;
    float sq = xv.x * xv.x + xv.y * xv.y + xv.z * xv.z + xv.w * xv.w;
    s  = wave_sum(s);
    sq = wave_sum(sq);
    const float mu  = s * (1.0f / CCH);
    const float var = sq * (1.0f / CCH) - mu * mu;
    const float rs  = rsqrtf(var + EPSV);
    unsigned short* dst = At + row * K1_LD + lane * 4;
    dst[0] = f2bf((xv.x - mu) * rs * ga.x + be.x);
    dst[1] = f2bf((xv.y - mu) * rs * ga.y + be.y);
    dst[2] = f2bf((xv.z - mu) * rs * ga.z + be.z);
    dst[3] = f2bf((xv.w - mu) * rs * ga.w + be.w);
  }
  __syncthreads();

  // GEMM: 4 row-subtiles x 33 col-tiles, K=128 (4 WMMAs each).
  for (int job = wid; job < 4 * 33; job += 8) {
    const int rt = job / 33;
    const int ct = job - rt * 33;
    const int col0 = ct * 16;
    v8f acc = {0.f, 0.f, 0.f, 0.f, 0.f, 0.f, 0.f, 0.f};
#pragma unroll
    for (int kc = 0; kc < 4; ++kc) {
      v16bf a = frag_a(At, rt * 16, kc * 32, K1_LD, lane);
      v16bf b = frag_bT(Wt, col0, kc * 32, K1_LD, lane);
      acc = WMMA_BF16(a, b, acc);
    }
    const int n = lane & 15, mhi = (lane >> 4) * 8;
    if (ct < 24) {                       // q / k / v -> [i][h][j][c] bf16
      const int which = ct >> 3;
      unsigned short* dst = (which == 0) ? qT : (which == 1) ? kT : vT;
      const int f = col0 - which * 128 + n;
      const int c = f >> 2, h = f & 3;   // f = c*H + h
#pragma unroll
      for (int r = 0; r < 8; ++r) {
        const int m = m0 + rt * 16 + mhi + r;
        const int i = m >> 8, j = m & 255;
        dst[(((size_t)i * NH_ + h) * NRES + j) * HC_ + c] = f2bf(acc[r]);
      }
    } else if (ct < 32) {                // g: sigmoid(x@Wg + bg) -> bf16 [m][f]
      const int f = col0 - 384 + n;
      const float bias = bg[f];
#pragma unroll
      for (int r = 0; r < 8; ++r) {
        const int m = m0 + rt * 16 + mhi + r;
        const float z = acc[r] + bias;
        gT[(size_t)m * DM_ + f] = f2bf(1.0f / (1.0f + __expf(-z)));
      }
    } else if (n < 4) {                  // b -> f32 bT[h][j][i]
#pragma unroll
      for (int r = 0; r < 8; ++r) {
        const int m = m0 + rt * 16 + mhi + r;
        const int i = m >> 8, j = m & 255;
        bT[((size_t)n * NRES + j) * NRES + i] = acc[r];
      }
    }
  }
}

// ---------------------------------------------------------------------------
// Kernel 2: attention per (i, h, j-tile of 64).
//   S = (Q K^T)/sqrt(HC) + b[k,j,h]; softmax over k (256); O = P V; O *= g.
// ---------------------------------------------------------------------------
__global__ void __launch_bounds__(256) k2_attn(
    const unsigned short* __restrict__ qT, const unsigned short* __restrict__ kT,
    const unsigned short* __restrict__ vT, const unsigned short* __restrict__ gT,
    const float* __restrict__ bT, unsigned short* __restrict__ aO)
{
  extern __shared__ unsigned char smem_raw[];
  unsigned short* Qs = (unsigned short*)smem_raw;     // 64 x 40 bf16
  unsigned short* Ks = Qs + 64 * 40;                  // 256 x 40 bf16
  unsigned short* Vt = Ks + 256 * 40;                 // 32 x 264 bf16 (transposed)
  float*          S  = (float*)(Vt + 32 * 264);       // 64 x 264 f32
  unsigned short* P  = (unsigned short*)(S + 64 * 264); // 64 x 264 bf16

  const int tid = threadIdx.x, lane = tid & 31, wid = tid >> 5;
  const int bid = blockIdx.x;
  const int i = bid >> 4, h = (bid >> 2) & 3, jt = bid & 3;
  const int j0 = jt * 64;

  const unsigned short* qbp = qT + ((size_t)(i * NH_ + h) * NRES) * HC_;
  const unsigned short* kbp = kT + ((size_t)(i * NH_ + h) * NRES) * HC_;
  const unsigned short* vbp = vT + ((size_t)(i * NH_ + h) * NRES) * HC_;

  { // Q tile: 64 rows x 32 (16B chunks, 256 threads exactly)
    const int row = tid >> 2, ch = tid & 3;
    *(uint4*)(Qs + row * 40 + ch * 8) = *(const uint4*)(qbp + (j0 + row) * HC_ + ch * 8);
  }
  { // K: 256 rows x 32
#pragma unroll
    for (int ch = 0; ch < 4; ++ch)
      *(uint4*)(Ks + tid * 40 + ch * 8) = *(const uint4*)(kbp + tid * HC_ + ch * 8);
  }
  // V transposed into Vt[c][k]
  for (int idx = tid; idx < NRES * HC_; idx += 256) {
    const int k = idx >> 5, c = idx & 31;
    Vt[c * 264 + k] = vbp[k * HC_ + c];
  }
  __syncthreads();

  // GEMM1: one WMMA per 16x16 score tile (K = HC = 32).
  const float sc = 0.17677669529663687f; // 1/sqrt(32)
  for (int job = wid; job < 64; job += 8) {
    const int rt = job >> 4, ct = job & 15;
    v16bf a = frag_a(Qs, rt * 16, 0, 40, lane);
    v16bf b = frag_bT(Ks, ct * 16, 0, 40, lane);
    v8f acc = {0.f, 0.f, 0.f, 0.f, 0.f, 0.f, 0.f, 0.f};
    acc = WMMA_BF16(a, b, acc);
    const int n = lane & 15, mhi = (lane >> 4) * 8;
#pragma unroll
    for (int r = 0; r < 8; ++r) {
      const int jl = rt * 16 + mhi + r;
      const int kc = ct * 16 + n;
      const float bias = bT[((size_t)h * NRES + (j0 + jl)) * NRES + kc]; // b[k, j, h]
      S[jl * 264 + kc] = acc[r] * sc + bias;
    }
  }
  __syncthreads();

  // Softmax over k (full 256 in LDS): one wave per row.
#pragma unroll
  for (int r8 = 0; r8 < 8; ++r8) {
    const int row = wid * 8 + r8;
    float* sp = S + row * 264 + lane * 8;
    const float4 a0 = *(const float4*)(sp);
    const float4 a1 = *(const float4*)(sp + 4);
    float mx = fmaxf(fmaxf(fmaxf(a0.x, a0.y), fmaxf(a0.z, a0.w)),
                     fmaxf(fmaxf(a1.x, a1.y), fmaxf(a1.z, a1.w)));
    mx = wave_max(mx);
    float e0 = __expf(a0.x - mx), e1 = __expf(a0.y - mx);
    float e2 = __expf(a0.z - mx), e3 = __expf(a0.w - mx);
    float e4 = __expf(a1.x - mx), e5 = __expf(a1.y - mx);
    float e6 = __expf(a1.z - mx), e7 = __expf(a1.w - mx);
    float s = wave_sum(e0 + e1 + e2 + e3 + e4 + e5 + e6 + e7);
    const float inv = 1.0f / s;
    alignas(16) unsigned short pk[8] = {
      f2bf(e0 * inv), f2bf(e1 * inv), f2bf(e2 * inv), f2bf(e3 * inv),
      f2bf(e4 * inv), f2bf(e5 * inv), f2bf(e6 * inv), f2bf(e7 * inv)};
    *(uint4*)(P + row * 264 + lane * 8) = *(const uint4*)pk;
  }
  __syncthreads();

  // GEMM2: O = P @ V (K = 256, 8 WMMAs), gate, store bf16 [m][f].
  {
    const int rt = wid >> 1, ct = wid & 1, c0 = ct * 16;
    v8f acc = {0.f, 0.f, 0.f, 0.f, 0.f, 0.f, 0.f, 0.f};
#pragma unroll
    for (int kc = 0; kc < 8; ++kc) {
      v16bf a = frag_a(P, rt * 16, kc * 32, 264, lane);
      v16bf b = frag_bT(Vt, c0, kc * 32, 264, lane);
      acc = WMMA_BF16(a, b, acc);
    }
    const int n = lane & 15, mhi = (lane >> 4) * 8;
#pragma unroll
    for (int r = 0; r < 8; ++r) {
      const int jl = rt * 16 + mhi + r;
      const int c  = c0 + n;
      const int mg = i * NRES + j0 + jl;
      const int f  = (c << 2) | h;                  // f = c*H + h
      const float gv = bf2f(gT[(size_t)mg * DM_ + f]);
      aO[(size_t)mg * DM_ + f] = f2bf(acc[r] * gv);
    }
  }
}

// ---------------------------------------------------------------------------
// Kernel 3: output projection  out = aO @ Wo + bo  (bf16 WMMA, fp32 out)
// ---------------------------------------------------------------------------
__global__ void __launch_bounds__(256) k3_outproj(
    const unsigned short* __restrict__ aO, const float* __restrict__ Wo,
    const float* __restrict__ bo, float* __restrict__ out)
{
  extern __shared__ unsigned char smem_raw[];
  unsigned short* At  = (unsigned short*)smem_raw;  // 64 x 136 bf16
  unsigned short* Wot = At + 64 * 136;              // 128 x 136 bf16, [col][k]

  const int tid = threadIdx.x, lane = tid & 31, wid = tid >> 5;
  const int m0 = blockIdx.x * 64;

  for (int idx = tid; idx < 128 * 128; idx += 256) {
    const int k = idx >> 7, f = idx & 127;
    Wot[f * 136 + k] = f2bf(Wo[k * 128 + f]);
  }
  for (int idx = tid; idx < 64 * 16; idx += 256) { // 16B chunks of the A tile
    const int row = idx >> 4, c8 = idx & 15;
    *(uint4*)(At + row * 136 + c8 * 8) =
        *(const uint4*)(aO + (size_t)(m0 + row) * 128 + c8 * 8);
  }
  __syncthreads();

  for (int job = wid; job < 32; job += 8) {
    const int rt = job >> 3, ct = job & 7;
    const int col0 = ct * 16;
    v8f acc = {0.f, 0.f, 0.f, 0.f, 0.f, 0.f, 0.f, 0.f};
#pragma unroll
    for (int kc = 0; kc < 4; ++kc) {
      v16bf a = frag_a(At, rt * 16, kc * 32, 136, lane);
      v16bf b = frag_bT(Wot, col0, kc * 32, 136, lane);
      acc = WMMA_BF16(a, b, acc);
    }
    const int n = lane & 15, mhi = (lane >> 4) * 8;
    const int col = col0 + n;
    const float bias = bo[col];
#pragma unroll
    for (int r = 0; r < 8; ++r) {
      const int m = m0 + rt * 16 + mhi + r;
      out[(size_t)m * 128 + col] = acc[r] + bias;
    }
  }
}

// ---------------------------------------------------------------------------
extern "C" void kernel_launch(void* const* d_in, const int* in_sizes, int n_in,
                              void* d_out, int out_size, void* d_ws, size_t ws_size,
                              hipStream_t stream) {
  (void)in_sizes; (void)n_in; (void)out_size; (void)ws_size;
  const float* x2d   = (const float*)d_in[0];
  const float* gamma = (const float*)d_in[1];
  const float* beta  = (const float*)d_in[2];
  const float* Wq    = (const float*)d_in[3];
  const float* Wk    = (const float*)d_in[4];
  const float* Wv    = (const float*)d_in[5];
  const float* Wb    = (const float*)d_in[6];
  const float* Wg    = (const float*)d_in[7];
  const float* bg    = (const float*)d_in[8];
  const float* Wo    = (const float*)d_in[9];
  const float* bo    = (const float*)d_in[10];

  const size_t MB = (size_t)1 << 20;
  unsigned char* ws = (unsigned char*)d_ws;
  unsigned short* qT = (unsigned short*)(ws);            // 16 MB bf16 [i][h][j][c]
  unsigned short* kT = (unsigned short*)(ws + 16 * MB);  // 16 MB
  unsigned short* vT = (unsigned short*)(ws + 32 * MB);  // 16 MB
  unsigned short* gT = (unsigned short*)(ws + 48 * MB);  // 16 MB bf16 [m][f]
  float*          bT = (float*)        (ws + 64 * MB);   //  1 MB f32 [h][j][i]
  unsigned short* aO = (unsigned short*)(ws + 65 * MB);  // 16 MB bf16 [m][f]

  const size_t s1 = (size_t)(64 * K1_LD + K1_NCOL * K1_LD) * 2;               // 161,024 B
  const size_t s2 = (size_t)(64*40 + 256*40 + 32*264) * 2 + (size_t)64*264*4
                    + (size_t)64*264*2;                                       // 143,872 B
  const size_t s3 = (size_t)(64 * 136 + 128 * 136) * 2;                       //  52,224 B

  k1_ln_proj<<<dim3(1024), dim3(256), s1, stream>>>(x2d, gamma, beta, Wq, Wk, Wv,
                                                    Wb, Wg, bg, qT, kT, vT, gT, bT);
  k2_attn  <<<dim3(4096), dim3(256), s2, stream>>>(qT, kT, vT, gT, bT, aO);
  k3_outproj<<<dim3(1024), dim3(256), s3, stream>>>(aO, Wo, bo, (float*)d_out);
}
